// UpgradedHSTGNN_22368189677720
// MI455X (gfx1250) — matrix operand order repaired and checked by reading, not compiled, not run on hardware
//
#include <hip/hip_runtime.h>
#include <math.h>

// ---------------------------------------------------------------------------
// CDNA5 (gfx1250) implementation of the hierarchical ST-GNN forward pass.
// All dense GEMMs run through v_wmma_f32_16x16x32_bf16 (wave32 WMMA),
// f32 activations converted to bf16 on the fly, f32 accumulation.
// Node dims padded (75->80, 4->8) so batched attention/graph GEMMs stay on
// the wide b128 staging path.
// ---------------------------------------------------------------------------

typedef __attribute__((ext_vector_type(16))) __bf16 v16bf;
typedef __attribute__((ext_vector_type(8)))  __bf16 v8bf;
typedef __attribute__((ext_vector_type(8)))  float  v8f;

#define DEV static __device__ __forceinline__

DEV unsigned short f2bf(float f) {
  union { float f; unsigned u; } c; c.f = f;
  unsigned u = c.u;
  return (unsigned short)((u + 0x7FFFu + ((u >> 16) & 1u)) >> 16);  // RNE
}

DEV unsigned pack2bf(float a, float b) {
  return (unsigned)f2bf(a) | ((unsigned)f2bf(b) << 16);
}

DEV float gelu_exact(float x) {
  return 0.5f * x * (1.0f + erff(x * 0.70710678118654752440f));
}

// --------------------------- WMMA GEMM -------------------------------------
// C[M,N] = epilogue(A[M,K](f32) * Bt[N,K](bf16)^T + bias) per batch z.
// Batched offsets: off = (z/zdiv)*s0 + (z%zdiv)*s1  (elements).
#define BM 128
#define BN 64
#define BK 32
#define GEMM_THREADS 256

enum { EP_NONE = 0, EP_GELU = 1, EP_RES = 2, EP_GELU_RES = 3, EP_GATE = 4 };

__global__ __launch_bounds__(GEMM_THREADS)
void k_wmma_gemm(const float* __restrict__ A, const unsigned short* __restrict__ BtU,
                 float* __restrict__ C, const float* __restrict__ bias,
                 const float* __restrict__ res0, const float* __restrict__ res1,
                 int M, int N, int K, int lda, int ldb, int ldc, int zdiv,
                 long long a_s0, long long a_s1, long long b_s0, long long b_s1,
                 long long c_s0, long long c_s1, int epilogue)
{
  __shared__ __align__(16) unsigned short As[BM][BK + 8];   // +8 pad: conflict-free b128
  __shared__ __align__(16) unsigned short Bs[BN][BK + 8];

  const int tid  = threadIdx.x;
  const int lane = tid & 31;
  const int wid  = tid >> 5;
  const int wm   = (wid & 3) * 32;   // wave M offset inside tile (4 waves in M)
  const int wn   = (wid >> 2) * 32;  // wave N offset inside tile (2 waves in N)
  const int half = lane >> 4;        // lane group 0/1
  const int l16  = lane & 15;

  const int z = blockIdx.z;
  const long long aoff = (long long)(z / zdiv) * a_s0 + (long long)(z % zdiv) * a_s1;
  const long long boff = (long long)(z / zdiv) * b_s0 + (long long)(z % zdiv) * b_s1;
  const long long coff = (long long)(z / zdiv) * c_s0 + (long long)(z % zdiv) * c_s1;

  const int m0 = blockIdx.y * BM;
  const int n0 = blockIdx.x * BN;

  // Alignment-only fast-path predicates (block-uniform). Out-of-range rows are
  // CLAMPED in the wide path: WMMA output row m depends only on A row m (and
  // col n only on B row n), so clamped/duplicated rows feed only elements that
  // the store mask discards.
  const bool alignedA = ((lda & 3) == 0) && ((aoff & 3) == 0);
  const bool alignedB = ((ldb & 7) == 0) && ((boff & 7) == 0);

  v8f acc[2][2];
#pragma unroll
  for (int i = 0; i < 2; i++)
#pragma unroll
    for (int j = 0; j < 2; j++) {
      v8f zz = {0.f, 0.f, 0.f, 0.f, 0.f, 0.f, 0.f, 0.f};
      acc[i][j] = zz;
    }

  for (int k0 = 0; k0 < K; k0 += BK) {
    const bool fullK = (k0 + BK <= K);

    // ---- stage A tile (f32 -> bf16) ----
    if (alignedA && fullK) {
      // wide path: b128 loads (row-clamped), packed bf16, ds_store_b64
      const float* Abase = A + aoff + k0;
      const int fcol = (tid & 7) * 4;
#pragma unroll
      for (int p = 0; p < 4; p++) {
        int r = (tid >> 3) + p * 32;
        int gr = m0 + r; gr = (gr < M) ? gr : (M - 1);
        const float4 v = *reinterpret_cast<const float4*>(Abase + (long long)gr * lda + fcol);
        uint2 u;
        u.x = pack2bf(v.x, v.y);
        u.y = pack2bf(v.z, v.w);
        *reinterpret_cast<uint2*>(&As[r][fcol]) = u;
      }
      // speculative prefetch of next A K-tile (global_prefetch_b8)
      if (k0 + BK < K) {
        int pr = m0 + (tid >> 1); pr = (pr < M) ? pr : (M - 1);
        __builtin_prefetch(Abase + (long long)pr * lda + BK, 0, 1);
      }
    } else {
      // edge path: clamped address (always valid) + multiplicative mask keeps
      // the load unconditional (no exec-branch, loads batch up), fully unrolled
#pragma unroll
      for (int t2 = 0; t2 < (BM * BK) / GEMM_THREADS; t2++) {
        int idx = tid + t2 * GEMM_THREADS;
        int r = idx >> 5, c = idx & 31;
        int gr = m0 + r, gk = k0 + c;
        int cr = (gr < M) ? gr : (M - 1);
        int ck = (gk < K) ? gk : (K - 1);
        float msk = (gr < M && gk < K) ? 1.0f : 0.0f;
        float v = A[aoff + (long long)cr * lda + ck];
        As[r][c] = f2bf(v * msk);
      }
    }

    // ---- stage Bt tile (bf16 already) ----
    if (alignedB && fullK) {
      const unsigned short* Bbase = BtU + boff + k0;
      int r = tid >> 2;
      int gn = n0 + r; gn = (gn < N) ? gn : (N - 1);
      int fcol = (tid & 3) * 8;
      uint4 v = *reinterpret_cast<const uint4*>(Bbase + (long long)gn * ldb + fcol);
      *reinterpret_cast<uint4*>(&Bs[r][fcol]) = v;
    } else {
#pragma unroll
      for (int t2 = 0; t2 < (BN * BK) / GEMM_THREADS; t2++) {
        int idx = tid + t2 * GEMM_THREADS;
        int r = idx >> 5, c = idx & 31;
        int gn = n0 + r, gk = k0 + c;
        int cn = (gn < N) ? gn : (N - 1);
        int ck = (gk < K) ? gk : (K - 1);
        unsigned m2 = (gn < N && gk < K) ? 0xFFFFu : 0u;
        unsigned v = BtU[boff + (long long)cn * ldb + ck];
        Bs[r][c] = (unsigned short)(v & m2);
      }
    }
    __syncthreads();

    // A fragments: 16-bit A 16x32 layout (lanes 0-15: K {0..7,16..23})
    v16bf aF[2], bF[2];
#pragma unroll
    for (int i = 0; i < 2; i++) {
      int row = wm + i * 16 + l16;
      v8bf lo = *reinterpret_cast<const v8bf*>(&As[row][half * 8]);
      v8bf hi = *reinterpret_cast<const v8bf*>(&As[row][half * 8 + 16]);
      aF[i] = __builtin_shufflevector(lo, hi, 0, 1, 2, 3, 4, 5, 6, 7,
                                      8, 9, 10, 11, 12, 13, 14, 15);
    }
    // B fragments: lane = column, lanes 0-15 K {0..7,16..23}, lanes 16-31 K {8..15,24..31}
#pragma unroll
    for (int j = 0; j < 2; j++) {
      int row = wn + j * 16 + l16;
      v8bf lo = *reinterpret_cast<const v8bf*>(&Bs[row][half * 8]);
      v8bf hi = *reinterpret_cast<const v8bf*>(&Bs[row][half * 8 + 16]);
      bF[j] = __builtin_shufflevector(lo, hi, 0, 1, 2, 3, 4, 5, 6, 7,
                                      8, 9, 10, 11, 12, 13, 14, 15);
    }
#pragma unroll
    for (int i = 0; i < 2; i++)
#pragma unroll
      for (int j = 0; j < 2; j++)
        acc[i][j] = __builtin_amdgcn_wmma_f32_16x16x32_bf16(
            false, aF[i], false, bF[j], (short)0, acc[i][j], false, false);
    __syncthreads();
  }

  // epilogue + store. C/D layout: VGPR r -> M = r + half*8, lane&15 -> N.
#pragma unroll
  for (int i = 0; i < 2; i++) {
#pragma unroll
    for (int j = 0; j < 2; j++) {
#pragma unroll
      for (int r = 0; r < 8; r++) {
        int gr = m0 + wm + i * 16 + half * 8 + r;
        int gc = n0 + wn + j * 16 + l16;
        if (gr < M && gc < N) {
          float v = acc[i][j][r];
          if (bias) v += bias[gc];
          long long ci = coff + (long long)gr * ldc + gc;
          if (epilogue == EP_GELU)            v = gelu_exact(v);
          else if (epilogue == EP_RES)        v += res0[ci];
          else if (epilogue == EP_GELU_RES)   v = res0[ci] + gelu_exact(v);
          else if (epilogue == EP_GATE) {
            float s = 1.0f / (1.0f + __expf(-v));
            v = res0[ci] + s * res1[ci];
          }
          C[ci] = v;
        }
      }
    }
  }
}

// ------------------- transpose + cast f32 -> bf16 --------------------------
// in [z][R][C] f32 -> out [z][C][ldOut] bf16 (pad cols ldOut>=R left untouched)
__global__ __launch_bounds__(256)
void k_transpose_bf16(const float* __restrict__ in, unsigned short* __restrict__ out,
                      int R, int Ccols, long long inBatch, long long outBatch, int ldOut)
{
  __shared__ float t[32][33];
  const int z = blockIdx.z;
  const float* ip = in + (long long)z * inBatch;
  unsigned short* op = out + (long long)z * outBatch;
  int r0 = blockIdx.y * 32, c0 = blockIdx.x * 32;
  int tx = threadIdx.x & 31, ty = threadIdx.x >> 5;
  for (int rr = ty; rr < 32; rr += 8) {
    int r = r0 + rr, c = c0 + tx;
    int cr = (r < R) ? r : (R - 1);
    int cc2 = (c < Ccols) ? c : (Ccols - 1);
    float msk = (r < R && c < Ccols) ? 1.0f : 0.0f;
    t[rr][tx] = ip[(long long)cr * Ccols + cc2] * msk;
  }
  __syncthreads();
  for (int cc = ty; cc < 32; cc += 8) {
    int c = c0 + cc, r = r0 + tx;
    if (c < Ccols && r < R) op[(long long)c * ldOut + r] = f2bf(t[tx][cc]);
  }
}

// ----------------------------- LayerNorm (D=256) ---------------------------
__global__ __launch_bounds__(256)
void k_layernorm(const float* __restrict__ in, float* __restrict__ out,
                 const float* __restrict__ g, const float* __restrict__ b,
                 int Mrows, int ldin, int ldout, int doGelu)
{
  int lane = threadIdx.x & 31;
  long long row = (long long)blockIdx.x * 8 + (threadIdx.x >> 5);
  if (row >= Mrows) return;
  const float* ip = in + row * ldin;
  float v[8];
  float s = 0.f;
#pragma unroll
  for (int i = 0; i < 8; i++) { v[i] = ip[lane + i * 32]; s += v[i]; }
#pragma unroll
  for (int m = 16; m >= 1; m >>= 1) s += __shfl_xor(s, m, 32);
  float mean = s * (1.0f / 256.0f);
  float q = 0.f;
#pragma unroll
  for (int i = 0; i < 8; i++) { float d = v[i] - mean; q += d * d; }
#pragma unroll
  for (int m = 16; m >= 1; m >>= 1) q += __shfl_xor(q, m, 32);
  float inv = rsqrtf(q * (1.0f / 256.0f) + 1e-5f);
  float* op = out + row * ldout;
#pragma unroll
  for (int i = 0; i < 8; i++) {
    int c = lane + i * 32;
    float y = (v[i] - mean) * inv * g[c] + b[c];
    if (doGelu) y = gelu_exact(y);
    op[c] = y;
  }
}

// ---------------------- row softmax (wave per row) -------------------------
// S rows with stride ld, valid width Ncol; optional bias pb [Rr][Ncol].
__global__ __launch_bounds__(256)
void k_softmax(float* __restrict__ S, const float* __restrict__ pb,
               long long nrowsTot, int Rr, int Ncol, int ld, float scale)
{
  int lane = threadIdx.x & 31;
  long long row = (long long)blockIdx.x * 8 + (threadIdx.x >> 5);
  if (row >= nrowsTot) return;
  int r = (int)(row % Rr);
  float* p = S + row * ld;
  const float* pbr = pb ? pb + (long long)r * Ncol : nullptr;
  float vv[4];
  int cnt = 0;
  float mx = -3.0e38f;
  for (int c = lane; c < Ncol; c += 32) {
    float v = p[c] * scale + (pbr ? pbr[c] : 0.0f);
    vv[cnt++] = v;
    mx = fmaxf(mx, v);
  }
#pragma unroll
  for (int m = 16; m >= 1; m >>= 1) mx = fmaxf(mx, __shfl_xor(mx, m, 32));
  float sum = 0.f;
  cnt = 0;
  for (int c = lane; c < Ncol; c += 32) { float e = __expf(vv[cnt] - mx); vv[cnt++] = e; sum += e; }
#pragma unroll
  for (int m = 16; m >= 1; m >>= 1) sum += __shfl_xor(sum, m, 32);
  float inv = 1.0f / sum;
  cnt = 0;
  for (int c = lane; c < Ncol; c += 32) p[c] = vv[cnt++] * inv;
}

// ------------------------- misc small kernels ------------------------------
// Am[z][rowElems] = mean over 4 heads of S[(z*4+h)][rowElems]
__global__ void k_mean_heads(const float* __restrict__ S, float* __restrict__ Am,
                             long long tot, int rowElems)
{
  long long idx = (long long)blockIdx.x * 256 + threadIdx.x;
  if (idx >= tot) return;
  long long z = idx / rowElems;
  int rm = (int)(idx % rowElems);
  const float* base = S + z * 4 * (long long)rowElems + rm;
  Am[idx] = 0.25f * (base[0] + base[rowElems] + base[2 * (long long)rowElems]
                     + base[3 * (long long)rowElems]);
}

__global__ void k_concat_fine(const float* __restrict__ pose, const float* __restrict__ lh,
                              const float* __restrict__ rh, float* __restrict__ x,
                              int bt0, int CBc)
{
  long long tot = (long long)CBc * 75 * 256;
  long long idx = (long long)blockIdx.x * 256 + threadIdx.x;
  if (idx >= tot) return;
  int d = (int)(idx & 255);
  long long t = idx >> 8;
  int n = (int)(t % 75);
  long long bt = bt0 + t / 75;
  float v;
  if (n < 33)      v = pose[(bt * 33 + n) * 256 + d];
  else if (n < 54) v = lh[(bt * 21 + (n - 33)) * 256 + d];
  else             v = rh[(bt * 21 + (n - 54)) * 256 + d];
  x[idx] = v;
}

// out[btl][h][n][64] bf16 = in[(btl*Nn+n)*ldin + colOff + h*64 + d]
__global__ void k_pack_k(const float* __restrict__ in, unsigned short* __restrict__ out,
                         int CBc, int Nn, int ldin, int colOff)
{
  long long tot = (long long)CBc * 4 * Nn * 64;
  long long idx = (long long)blockIdx.x * 256 + threadIdx.x;
  if (idx >= tot) return;
  int d = (int)(idx & 63);
  long long t = idx >> 6;
  int n = (int)(t % Nn); t /= Nn;
  int h = (int)(t & 3);
  long long btl = t >> 2;
  out[idx] = f2bf(in[(btl * Nn + n) * (long long)ldin + colOff + h * 64 + d]);
}

// out[btl][h][d][NnP] bf16 = V^T (zero-padded to NnP columns)
__global__ void k_pack_vt(const float* __restrict__ in, unsigned short* __restrict__ out,
                          int CBc, int Nn, int NnP, int ldin, int colOff)
{
  long long tot = (long long)CBc * 4 * 64 * NnP;
  long long idx = (long long)blockIdx.x * 256 + threadIdx.x;
  if (idx >= tot) return;
  int n = (int)(idx % NnP);
  long long t = idx / NnP;
  int d = (int)(t & 63); t >>= 6;
  int h = (int)(t & 3);
  long long btl = t >> 2;
  int cn = (n < Nn) ? n : (Nn - 1);
  float msk = (n < Nn) ? 1.0f : 0.0f;
  float v = in[(btl * Nn + cn) * (long long)ldin + colOff + h * 64 + d];
  out[idx] = f2bf(v * msk);
}

__global__ void k_mean_rows(const float* __restrict__ in, float* __restrict__ out,
                            int BTtot, int Nn, int s0, int len)
{
  long long tot = (long long)BTtot * 256;
  long long idx = (long long)blockIdx.x * 256 + threadIdx.x;
  if (idx >= tot) return;
  int d = (int)(idx & 255);
  long long bt = idx >> 8;
  float a = 0.f;
  for (int n = 0; n < len; n++) a += in[(bt * Nn + s0 + n) * 256 + d];
  out[idx] = a / (float)len;
}

__global__ void k_fuse(const float* __restrict__ fine, const float* __restrict__ high,
                       const float* __restrict__ face, float* __restrict__ out)
{
  long long idx = (long long)blockIdx.x * 256 + threadIdx.x;
  if (idx >= (long long)1024 * 256) return;
  int d = (int)(idx & 255);
  long long bt = idx >> 8;
  float s = 0.f;
  for (int n = 0; n < 75; n++) s += fine[(bt * 75 + n) * 256 + d];
  float s2 = 0.f;
  for (int j = 0; j < 4; j++) s2 += high[(bt * 4 + j) * 256 + d];
  out[idx] = s * (1.0f / 75.0f) + s2 * 0.25f + face[idx];
}

// ---------------------------------------------------------------------------
// Host orchestration
// ---------------------------------------------------------------------------
// Param leaf indices assume recursive insertion-order flattening of the
// setup_inputs() dict:
//   0..3: x_pose, x_lhand, x_rhand, x_raw
//   4: fine_adj {wq,bq,wk,bk,pb}
//   9: fine_gcn[2] {g,b,w,bi}
//  17: fine_tr[2] {g1,b1,g2,b2,w_in,b_in,w_out,b_out,w1,bf1,w2,bf2,wg,bg}
//  45: high_adj, 50: high_gcn[2], 58: high_tr[2]
//  86: face {w,b,g,be}, 90: pool {pose,left,right,face}x{w,b,g,be}, 106: out

struct StreamW {
  unsigned short *wqT, *wkT;
  unsigned short *gwT[2];
  unsigned short *winT[2], *wgT[2], *woutT[2], *w1T[2], *w2T[2];
  int adjBase;
  int gcnBase[2];
  int trBase[2];
};

extern "C" void kernel_launch(void* const* d_in, const int* in_sizes, int n_in,
                              void* d_out, int out_size, void* d_ws, size_t ws_size,
                              hipStream_t stream)
{
  (void)in_sizes; (void)n_in; (void)out_size;
  const float* x_pose = (const float*)d_in[0];
  const float* x_lh   = (const float*)d_in[1];
  const float* x_rh   = (const float*)d_in[2];
  const float* x_raw  = (const float*)d_in[3];
  auto P = [&](int i) { return (const float*)d_in[i]; };

  const int CB = 128, NCH = 8;           // fine-stream batch chunking (8 x 128 bt)
  const int NP_F = 80, NP_H = 8;         // padded node dims (75->80, 4->8)

  // ---------------- workspace layout ----------------
  char* ws = (char*)d_ws;
  size_t off = 0;
  auto alloc = [&](size_t bytes) -> void* {
    off = (off + 255) & ~(size_t)255;
    void* p = (void*)(ws + off);
    off += bytes;
    return p;
  };

  StreamW FW, HW;
  FW.adjBase = 4;  FW.gcnBase[0] = 9;  FW.gcnBase[1] = 13; FW.trBase[0] = 17; FW.trBase[1] = 31;
  HW.adjBase = 45; HW.gcnBase[0] = 50; HW.gcnBase[1] = 54; HW.trBase[0] = 58; HW.trBase[1] = 72;

  auto allocW = [&](StreamW& S) {
    S.wqT = (unsigned short*)alloc(256 * 256 * 2);
    S.wkT = (unsigned short*)alloc(256 * 256 * 2);
    for (int l = 0; l < 2; l++) {
      S.gwT[l]   = (unsigned short*)alloc(256 * 256 * 2);
      S.winT[l]  = (unsigned short*)alloc(768 * 256 * 2);
      S.wgT[l]   = (unsigned short*)alloc(256 * 256 * 2);
      S.woutT[l] = (unsigned short*)alloc(256 * 256 * 2);
      S.w1T[l]   = (unsigned short*)alloc(1024 * 256 * 2);
      S.w2T[l]   = (unsigned short*)alloc(256 * 1024 * 2);
    }
  };
  allocW(FW);
  allocW(HW);
  unsigned short* faceWT = (unsigned short*)alloc((size_t)256 * 1872 * 2);
  unsigned short* poolWT[4];
  for (int j = 0; j < 4; j++) poolWT[j] = (unsigned short*)alloc(256 * 256 * 2);
  unsigned short* outWT = (unsigned short*)alloc(256 * 256 * 2);

  float* fineout = (float*)alloc((size_t)76800 * 256 * 4);   // persists (pooling/fuse)
  float* facebuf = (float*)alloc((size_t)1024 * 256 * 4);
  float* highx   = (float*)alloc((size_t)4096 * 256 * 4);
  float* pm      = (float*)alloc((size_t)1024 * 256 * 4);
  float* lm      = (float*)alloc((size_t)1024 * 256 * 4);
  float* rm      = (float*)alloc((size_t)1024 * 256 * 4);
  float* pooltmp = (float*)alloc((size_t)1024 * 256 * 4);
  float* fusedb  = (float*)alloc((size_t)1024 * 256 * 4);
  float* outtmp  = (float*)alloc((size_t)1024 * 256 * 4);

  size_t transBase = off;
  // fine-stream transients (per chunk of 128 bt -> 9600 rows)
  float* Am    = (float*)alloc((size_t)CB * 75 * NP_F * 4);
  float* hb    = (float*)alloc((size_t)CB * 75 * 256 * 4);
  float* ncb   = (float*)alloc((size_t)CB * 75 * 256 * 4);
  float* hgb   = (float*)alloc((size_t)CB * 75 * 256 * 4);
  float* ob    = (float*)alloc((size_t)CB * 75 * 256 * 4);
  float* qkvb  = (float*)alloc((size_t)CB * 75 * 768 * 4);
  float* ffnb  = (float*)alloc((size_t)CB * 75 * 1024 * 4);
  float* scb   = (float*)alloc((size_t)CB * 4 * 75 * NP_F * 4);
  unsigned short* htb = (unsigned short*)alloc((size_t)CB * 256 * NP_F * 2);
  unsigned short* kbb = (unsigned short*)alloc((size_t)CB * 4 * 75 * 64 * 2);
  unsigned short* vtb = (unsigned short*)alloc((size_t)CB * 4 * 64 * NP_F * 2);
  size_t fineEnd = off;

  // high-stream transients overlap the fine region (fine completes first)
  off = transBase;
  float* Amh   = (float*)alloc((size_t)1024 * 4 * NP_H * 4);
  float* hb_h  = (float*)alloc((size_t)4096 * 256 * 4);
  float* nc_h  = (float*)alloc((size_t)4096 * 256 * 4);
  float* hg_h  = (float*)alloc((size_t)4096 * 256 * 4);
  float* o_h   = (float*)alloc((size_t)4096 * 256 * 4);
  float* qkv_h = (float*)alloc((size_t)4096 * 768 * 4);
  float* ffn_h = (float*)alloc((size_t)4096 * 1024 * 4);
  float* sc_h  = (float*)alloc((size_t)1024 * 4 * 4 * NP_H * 4);
  unsigned short* ht_h = (unsigned short*)alloc((size_t)1024 * 256 * NP_H * 2);
  unsigned short* kb_h = (unsigned short*)alloc((size_t)1024 * 4 * 4 * 64 * 2);
  unsigned short* vt_h = (unsigned short*)alloc((size_t)1024 * 4 * 64 * NP_H * 2);
  size_t need = (fineEnd > off) ? fineEnd : off;
  if (need > ws_size) return;  // workspace too small; nothing safe to do

  // ---------------- launch helpers ----------------
  auto gemm = [&](const float* Ap, const unsigned short* Bt, float* Cp,
                  const float* biasp, const float* r0, const float* r1,
                  int M, int N, int K, int lda, int ldb, int ldc,
                  int zb, int zdiv,
                  long long as0, long long as1, long long bs0, long long bs1,
                  long long cs0, long long cs1, int ep) {
    dim3 g((N + BN - 1) / BN, (M + BM - 1) / BM, zb);
    k_wmma_gemm<<<g, GEMM_THREADS, 0, stream>>>(Ap, Bt, Cp, biasp, r0, r1,
        M, N, K, lda, ldb, ldc, zdiv, as0, as1, bs0, bs1, cs0, cs1, ep);
  };
  auto LN = [&](const float* in, float* out, const float* g, const float* b,
                int Mr, int ldin, int ldout, int gelu) {
    k_layernorm<<<dim3((Mr + 7) / 8), 256, 0, stream>>>(in, out, g, b, Mr, ldin, ldout, gelu);
  };
  auto TRN = [&](const float* in, unsigned short* out, int R, int C, int Z, int ldOut) {
    k_transpose_bf16<<<dim3((C + 31) / 32, (R + 31) / 32, Z), 256, 0, stream>>>(
        in, out, R, C, (long long)R * C, (long long)C * ldOut, ldOut);
  };
  auto SOFTMAX = [&](float* S, const float* pb, long long rows, int Rr, int Ncol, int ld, float sc) {
    k_softmax<<<dim3((unsigned)((rows + 7) / 8)), 256, 0, stream>>>(S, pb, rows, Rr, Ncol, ld, sc);
  };
  auto blocks1d = [](long long tot) { return dim3((unsigned)((tot + 255) / 256)); };
  auto PACKK = [&](const float* in, unsigned short* out, int CBc, int Nn, int ldin, int colOff) {
    k_pack_k<<<blocks1d((long long)CBc * 4 * Nn * 64), 256, 0, stream>>>(in, out, CBc, Nn, ldin, colOff);
  };
  auto PACKVT = [&](const float* in, unsigned short* out, int CBc, int Nn, int NnP,
                    int ldin, int colOff) {
    k_pack_vt<<<blocks1d((long long)CBc * 4 * 64 * NnP), 256, 0, stream>>>(
        in, out, CBc, Nn, NnP, ldin, colOff);
  };
  auto wtrans = [&](int pidx, unsigned short* dst, int K, int N) {
    k_transpose_bf16<<<dim3((N + 31) / 32, (K + 31) / 32, 1), 256, 0, stream>>>(
        P(pidx), dst, K, N, 0, 0, K);
  };

  // ---------------- weight pre-transposition (f32 [K,N] -> bf16 [N,K]) ------
  auto transW = [&](const StreamW& S) {
    wtrans(S.adjBase + 0, S.wqT, 256, 256);
    wtrans(S.adjBase + 2, S.wkT, 256, 256);
    for (int l = 0; l < 2; l++) {
      wtrans(S.gcnBase[l] + 2, S.gwT[l], 256, 256);
      int tB = S.trBase[l];
      wtrans(tB + 4,  S.winT[l],  256, 768);
      wtrans(tB + 12, S.wgT[l],   256, 256);
      wtrans(tB + 6,  S.woutT[l], 256, 256);
      wtrans(tB + 8,  S.w1T[l],   256, 1024);
      wtrans(tB + 10, S.w2T[l],   1024, 256);
    }
  };
  transW(FW);
  transW(HW);
  wtrans(86, faceWT, 1872, 256);
  for (int j = 0; j < 4; j++) wtrans(90 + j * 4, poolWT[j], 256, 256);
  wtrans(106, outWT, 256, 256);

  // ---------------- shared per-stream routines ------------------------------
  auto do_adj = [&](const float* xin, int CBc, int Nn, int NnP, const StreamW& S,
                    float* qb, float* kb2, unsigned short* kbfp, float* scp, float* AmP) {
    int Mr = CBc * Nn;
    long long rowE = (long long)Nn * NnP;    // score row block per head
    gemm(xin, S.wqT, qb, P(S.adjBase + 1), nullptr, nullptr, Mr, 256, 256,
         256, 256, 256, 1, 1, 0, 0, 0, 0, 0, 0, EP_NONE);
    gemm(xin, S.wkT, kb2, P(S.adjBase + 3), nullptr, nullptr, Mr, 256, 256,
         256, 256, 256, 1, 1, 0, 0, 0, 0, 0, 0, EP_NONE);
    PACKK(kb2, kbfp, CBc, Nn, 256, 0);
    gemm(qb, kbfp, scp, nullptr, nullptr, nullptr, Nn, Nn, 64, 256, 64, NnP,
         CBc * 4, 4, (long long)Nn * 256, 64, (long long)4 * Nn * 64,
         (long long)Nn * 64, 4 * rowE, rowE, EP_NONE);
    SOFTMAX(scp, P(S.adjBase + 4), (long long)CBc * 4 * Nn, Nn, Nn, NnP, 0.125f);
    k_mean_heads<<<blocks1d((long long)CBc * rowE), 256, 0, stream>>>(
        scp, AmP, (long long)CBc * rowE, (int)rowE);
  };

  auto do_layers = [&](float* x, const float* AmP, int CBc, int Nn, int NnP,
                       const StreamW& S,
                       float* hbP, float* ncP, float* hgP, float* qkvP, float* ffnP,
                       float* oP, float* scP, unsigned short* htP,
                       unsigned short* kbP, unsigned short* vtP) {
    int Mr = CBc * Nn;
    long long rowE = (long long)Nn * NnP;
    for (int li = 0; li < 2; li++) {
      // ---- GCN block: x += gelu((Am @ LN(x)) @ w + bi)
      int gB = S.gcnBase[li];
      LN(x, hbP, P(gB + 0), P(gB + 1), Mr, 256, 256, 0);
      TRN(hbP, htP, Nn, 256, CBc, NnP);
      gemm(AmP, htP, ncP, nullptr, nullptr, nullptr, Nn, 256, Nn, NnP, NnP, 256,
           CBc, 1, rowE, 0, (long long)256 * NnP, 0, (long long)Nn * 256, 0, EP_NONE);
      gemm(ncP, S.gwT[li], x, P(gB + 3), x, nullptr, Mr, 256, 256,
           256, 256, 256, 1, 1, 0, 0, 0, 0, 0, 0, EP_GELU_RES);
      // ---- transformer block
      int tB = S.trBase[li];
      LN(x, hbP, P(tB + 0), P(tB + 1), Mr, 256, 256, 0);
      TRN(hbP, htP, Nn, 256, CBc, NnP);
      gemm(AmP, htP, ncP, nullptr, nullptr, nullptr, Nn, 256, Nn, NnP, NnP, 256,
           CBc, 1, rowE, 0, (long long)256 * NnP, 0, (long long)Nn * 256, 0, EP_NONE);
      // hg = h + sigmoid(nc@wg + bg) * nc
      gemm(ncP, S.wgT[li], hgP, P(tB + 13), hbP, ncP, Mr, 256, 256,
           256, 256, 256, 1, 1, 0, 0, 0, 0, 0, 0, EP_GATE);
      // qkv
      gemm(hgP, S.winT[li], qkvP, P(tB + 5), nullptr, nullptr, Mr, 768, 256,
           256, 256, 768, 1, 1, 0, 0, 0, 0, 0, 0, EP_NONE);
      PACKK(qkvP, kbP, CBc, Nn, 768, 256);
      PACKVT(qkvP, vtP, CBc, Nn, NnP, 768, 512);
      // scores = Q K^T
      gemm(qkvP, kbP, scP, nullptr, nullptr, nullptr, Nn, Nn, 64, 768, 64, NnP,
           CBc * 4, 4, (long long)Nn * 768, 64, (long long)4 * Nn * 64,
           (long long)Nn * 64, 4 * rowE, rowE, EP_NONE);
      SOFTMAX(scP, nullptr, (long long)CBc * 4 * Nn, Nn, Nn, NnP, 0.125f);
      // o = P @ V   (written back interleaved by head)
      gemm(scP, vtP, oP, nullptr, nullptr, nullptr, Nn, 64, Nn, NnP, NnP, 256,
           CBc * 4, 4, 4 * rowE, rowE, (long long)4 * 64 * NnP, (long long)64 * NnP,
           (long long)Nn * 256, 64, EP_NONE);
      // x += o @ w_out + b_out
      gemm(oP, S.woutT[li], x, P(tB + 7), x, nullptr, Mr, 256, 256,
           256, 256, 256, 1, 1, 0, 0, 0, 0, 0, 0, EP_RES);
      // FFN
      LN(x, hbP, P(tB + 2), P(tB + 3), Mr, 256, 256, 0);
      gemm(hbP, S.w1T[li], ffnP, P(tB + 9), nullptr, nullptr, Mr, 1024, 256,
           256, 256, 1024, 1, 1, 0, 0, 0, 0, 0, 0, EP_GELU);
      gemm(ffnP, S.w2T[li], x, P(tB + 11), x, nullptr, Mr, 256, 1024,
           1024, 1024, 256, 1, 1, 0, 0, 0, 0, 0, 0, EP_RES);
    }
  };

  // ---------------- fine stream (chunked over bt) ----------------
  for (int ch = 0; ch < NCH; ch++) {
    int bt0 = ch * CB;
    float* x = fineout + (long long)bt0 * 75 * 256;
    k_concat_fine<<<blocks1d((long long)CB * 75 * 256), 256, 0, stream>>>(
        x_pose, x_lh, x_rh, x, bt0, CB);
    do_adj(x, CB, 75, NP_F, FW, hb, ncb, kbb, scb, Am);
    do_layers(x, Am, CB, 75, NP_F, FW, hb, ncb, hgb, qkvb, ffnb, ob, scb, htb, kbb, vtb);
  }

  // ---------------- face branch ----------------
  // face_raw = x_raw[:, :, 75:543, :] -> rows of 1872 f32, row stride 2172
  gemm(x_raw + 300, faceWT, pooltmp, P(87), nullptr, nullptr, 1024, 256, 1872,
       2172, 1872, 256, 1, 1, 0, 0, 0, 0, 0, 0, EP_NONE);
  LN(pooltmp, facebuf, P(88), P(89), 1024, 256, 256, 1);   // gelu(LN(...))

  // ---------------- pooling -> high-stream nodes ----------------
  k_mean_rows<<<blocks1d((long long)1024 * 256), 256, 0, stream>>>(fineout, pm, 1024, 75, 0, 33);
  k_mean_rows<<<blocks1d((long long)1024 * 256), 256, 0, stream>>>(fineout, lm, 1024, 75, 33, 21);
  k_mean_rows<<<blocks1d((long long)1024 * 256), 256, 0, stream>>>(fineout, rm, 1024, 75, 54, 21);
  const float* poolSrc[4] = { pm, lm, rm, facebuf };
  for (int j = 0; j < 4; j++) {
    int pB = 90 + j * 4;
    gemm(poolSrc[j], poolWT[j], pooltmp, P(pB + 1), nullptr, nullptr, 1024, 256, 256,
         256, 256, 256, 1, 1, 0, 0, 0, 0, 0, 0, EP_NONE);
    LN(pooltmp, highx + j * 256, P(pB + 2), P(pB + 3), 1024, 256, 1024, 0);
  }

  // ---------------- high stream (N=4, single pass) ----------------
  do_adj(highx, 1024, 4, NP_H, HW, hb_h, nc_h, kb_h, sc_h, Amh);
  do_layers(highx, Amh, 1024, 4, NP_H, HW, hb_h, nc_h, hg_h, qkv_h, ffn_h, o_h, sc_h,
            ht_h, kb_h, vt_h);

  // ---------------- fuse + output head ----------------
  k_fuse<<<blocks1d((long long)1024 * 256), 256, 0, stream>>>(fineout, highx, facebuf, fusedb);
  gemm(fusedb, outWT, outtmp, P(107), nullptr, nullptr, 1024, 256, 256,
       256, 256, 256, 1, 1, 0, 0, 0, 0, 0, 0, EP_NONE);
  LN(outtmp, (float*)d_out, P(108), P(109), 1024, 256, 256, 0);
}
